// CGC_66606352827237
// MI455X (gfx1250) — matrix operand order repaired and checked by reading, not compiled, not run on hardware
//
#include <hip/hip_runtime.h>
#include <hip/hip_bf16.h>
#include <cstdint>
#include <cstddef>

typedef __attribute__((ext_vector_type(16))) _Float16 v16h;
typedef __attribute__((ext_vector_type(8)))  _Float16 h8;
typedef __attribute__((ext_vector_type(8)))  float    v8f;
typedef int v4i __attribute__((vector_size(16)));   // matches async builtin param type

#if defined(__has_builtin)
#if __has_builtin(__builtin_amdgcn_global_load_async_to_lds_b128)
#define HAVE_ASYNC_LDS 1
#endif
#if __has_builtin(__builtin_amdgcn_s_barrier_signal) && __has_builtin(__builtin_amdgcn_s_barrier_wait)
#define HAVE_SPLIT_BARRIER 1
#endif
#endif
#ifndef HAVE_ASYNC_LDS
#define HAVE_ASYNC_LDS 0
#endif
#ifndef HAVE_SPLIT_BARRIER
#define HAVE_SPLIT_BARRIER 0
#endif

#define CGC_AS1 __attribute__((address_space(1)))
#define CGC_AS3 __attribute__((address_space(3)))

namespace {
constexpr int T    = 3;
constexpr int Bsz  = 8192;
constexpr int Dd   = 1024;
constexpr int Hh   = 512;
constexpr int Ee   = 256;
constexpr int NEXP = 16;        // 12 task-specific + 4 shared experts
constexpr int KC   = 64;        // K-chunk per LDS stage (2 WMMA K-steps)
constexpr int LDSB_STRIDE = 72; // f16 per column slot: 64 data + 8 pad (16B-aligned, conflict-free)
constexpr int LDSB_ELEMS  = 64 * LDSB_STRIDE;  // one buffer
}

// ---------------------------------------------------------------------------
// fp32 -> f16 elementwise convert
// ---------------------------------------------------------------------------
__global__ void cgc_f32_to_f16(const float* __restrict__ src,
                               _Float16* __restrict__ dst, size_t n) {
    size_t i = (size_t)blockIdx.x * blockDim.x + threadIdx.x;
    if (i < n) dst[i] = (_Float16)src[i];
}

// ---------------------------------------------------------------------------
// Weight convert + transpose: out[e][n][k] = src_e[k*N + n]   (fp32 -> f16)
// ---------------------------------------------------------------------------
__global__ void cgc_w_transpose(const float* __restrict__ wspec,
                                const float* __restrict__ wsh, int nspec,
                                _Float16* __restrict__ out, int K, int N) {
    size_t i   = (size_t)blockIdx.x * blockDim.x + threadIdx.x;
    size_t per = (size_t)K * N;
    if (i >= (size_t)NEXP * per) return;
    int    e = (int)(i / per);
    size_t r = i % per;
    int    n = (int)(r / K);
    int    k = (int)(r % K);
    const float* src = (e < nspec) ? (wspec + (size_t)e * per)
                                   : (wsh + (size_t)(e - nspec) * per);
    out[i] = (_Float16)src[(size_t)k * N + n];
}

// ---------------------------------------------------------------------------
// Cooperative B-tile stage: 64 columns x 64 K-values f16 into LDS.
// 256 threads, one 32B slice each (2x async b128). ASYNCcnt-tracked.
// ---------------------------------------------------------------------------
__device__ __forceinline__ void cgc_fill_b(const _Float16* __restrict__ Bt,
                                           _Float16* __restrict__ buf,
                                           int K, int n0, int kb) {
    const int tid  = (int)threadIdx.x;   // 0..255
    const int j    = tid >> 2;           // column 0..63
    const int part = tid & 3;            // 32B slice within 128B column chunk
    const _Float16* g = Bt + (size_t)(n0 + j) * K + kb + part * 16;
    _Float16*       l = buf + j * LDSB_STRIDE + part * 16;
#if HAVE_ASYNC_LDS
    // AS1/AS3 pointers built via integer round-trip: generic LDS pointers carry
    // the LDS byte offset in their low 32 bits (flat aperture mapping).
    __builtin_amdgcn_global_load_async_to_lds_b128(
        (CGC_AS1 v4i*)(uintptr_t)g,
        (CGC_AS3 v4i*)(uint32_t)(uintptr_t)l, 0, 0);
    __builtin_amdgcn_global_load_async_to_lds_b128(
        (CGC_AS1 v4i*)(uintptr_t)(g + 8),
        (CGC_AS3 v4i*)(uint32_t)(uintptr_t)(l + 8), 0, 0);
#else
    *(h8*)l = *(const h8*)g;
    *(h8*)(l + 8) = *(const h8*)(g + 8);
#endif
}

// Wait for this wave's async fills, then pure execution barrier (no full
// memory fence -> in-flight A-register loads are not forced to drain).
__device__ __forceinline__ void cgc_stage_commit() {
#if HAVE_ASYNC_LDS
    asm volatile("s_wait_asynccnt 0" ::: "memory");
#endif
#if HAVE_SPLIT_BARRIER && HAVE_ASYNC_LDS
    asm volatile("" ::: "memory");
    __builtin_amdgcn_s_barrier_signal(-1);
    __builtin_amdgcn_s_barrier_wait(-1);
    asm volatile("" ::: "memory");
#else
    __syncthreads();
#endif
}

// A-fragment register preload for one 64-wide K chunk (two 16x32 fragments).
__device__ __forceinline__ void cgc_load_a(h8 a[4], const _Float16* __restrict__ arow,
                                           int kc, int aoff) {
    a[0] = *(const h8*)(arow + kc + aoff);
    a[1] = *(const h8*)(arow + kc + aoff + 16);
    a[2] = *(const h8*)(arow + kc + 32 + aoff);
    a[3] = *(const h8*)(arow + kc + 32 + aoff + 16);
}

// 8 WMMAs on one staged 64x64 B chunk: 2 K-substeps x 4 N-tiles.
__device__ __forceinline__ void cgc_compute_chunk(
    const _Float16* __restrict__ bufB, const h8 a[4],
    v8f acc[4], int half, int ml)
{
#pragma unroll
    for (int sub = 0; sub < 2; ++sub) {
        v16h afrag = __builtin_shufflevector(a[2 * sub], a[2 * sub + 1],
            0, 1, 2, 3, 4, 5, 6, 7, 8, 9, 10, 11, 12, 13, 14, 15);
        const _Float16* bbase = bufB + sub * 32 + half * 8;
#pragma unroll
        for (int t = 0; t < 4; ++t) {
            const _Float16* bp = bbase + (t * 16 + ml) * LDSB_STRIDE;
            h8 blo = *(const h8*)bp;          // ds_load_b128
            h8 bhi = *(const h8*)(bp + 16);   // ds_load_b128
            v16h bfrag = __builtin_shufflevector(blo, bhi,
                0, 1, 2, 3, 4, 5, 6, 7, 8, 9, 10, 11, 12, 13, 14, 15);
            acc[t] = __builtin_amdgcn_wmma_f32_16x16x32_f16(
                false, afrag, false, bfrag, (short)0, acc[t], false, false);
        }
    }
}

// ---------------------------------------------------------------------------
// Wave-level 16(M) x 64(N) GEMM tile on v_wmma_f32_16x16x32_f16.
// B staged in double-buffered LDS; pipeline manually unrolled 2 chunks per
// loop body so buffer indices are compile-time (no acc/ptr phi copies).
// Requires K % 128 == 0 (true for K = 1024 and 512).
// ---------------------------------------------------------------------------
__device__ __forceinline__ void cgc_gemm_tile(
    const _Float16* __restrict__ A, const _Float16* __restrict__ Bt,
    const float* __restrict__ bias, _Float16* __restrict__ C,
    _Float16* __restrict__ Bs,     // [2 * LDSB_ELEMS] LDS
    int K, int N, int m0, int n0)
{
    const int lane = (int)(threadIdx.x & 31u);
    const int half = lane >> 4;    // 0 or 1
    const int ml   = lane & 15;
    const int aoff = half * 8;     // A: first 8-run K offset within 32-chunk

    const _Float16* arow = A + (size_t)(m0 + ml) * K;
    _Float16* buf0 = Bs;
    _Float16* buf1 = Bs + LDSB_ELEMS;

    v8f acc[4] = {};
    h8  a0[4], a1[4];

    // Prologue: stage chunk 0 into buf0, preload its A regs.
    cgc_fill_b(Bt, buf0, K, n0, 0);
    cgc_load_a(a0, arow, 0, aoff);
    cgc_stage_commit();

    const int nsteps = K / KC;     // even by construction
    for (int i = 0; i < nsteps; i += 2) {
        const int kc = i * KC;
        if (i + 1 < nsteps) {      // stage odd chunk while computing even one
            cgc_fill_b(Bt, buf1, K, n0, kc + KC);
            cgc_load_a(a1, arow, kc + KC, aoff);
            __builtin_prefetch(arow + kc + 2 * KC + aoff, 0, 3);  // global_prefetch_b8
        }
        cgc_compute_chunk(buf0, a0, acc, half, ml);
        cgc_stage_commit();

        if (i + 2 < nsteps) {      // stage next even chunk while computing odd
            cgc_fill_b(Bt, buf0, K, n0, kc + 2 * KC);
            cgc_load_a(a0, arow, kc + 2 * KC, aoff);
        }
        cgc_compute_chunk(buf1, a1, acc, half, ml);
        if (i + 2 < nsteps) cgc_stage_commit();
    }

    // C/D layout: VGPR r -> row m0 + r + 8*half, col n0 + 16*t + (lane&15)
    const int rbase = half * 8;
#pragma unroll
    for (int t = 0; t < 4; ++t) {
        const int   col = n0 + t * 16 + ml;
        const float bv  = bias[col];
#pragma unroll
        for (int r = 0; r < 8; ++r) {
            float v = acc[t][r] + bv;
            v = v > 0.f ? v : 0.f;
            C[(size_t)(m0 + rbase + r) * N + col] = (_Float16)v;
        }
    }
}

// Layer 1: h[e] = relu(X_e @ W1_e + b1_e),  X_e in {x_tasks[t], x_shared}
__global__ void __launch_bounds__(256)
cgc_layer1(const _Float16* __restrict__ xt, const _Float16* __restrict__ xs,
           const _Float16* __restrict__ w1t, const float* __restrict__ bspec,
           const float* __restrict__ bsh, _Float16* __restrict__ hout)
{
    __shared__ _Float16 Bs[2 * LDSB_ELEMS];
    const int e  = (int)blockIdx.z;
    const int w  = (int)(threadIdx.x >> 5);
    const int m0 = (int)blockIdx.y * 128 + w * 16;
    const int n0 = (int)blockIdx.x * 64;
    const _Float16* A    = (e < 12) ? (xt + (size_t)(e >> 2) * Bsz * Dd) : xs;
    const _Float16* Bt   = w1t + (size_t)e * Hh * Dd;
    const float*    bias = (e < 12) ? (bspec + (size_t)e * Hh)
                                    : (bsh + (size_t)(e - 12) * Hh);
    _Float16* C = hout + (size_t)e * Bsz * Hh;
    cgc_gemm_tile(A, Bt, bias, C, Bs, Dd, Hh, m0, n0);
}

// Layer 2: eo[e] = relu(h[e] @ W2_e + b2_e)
__global__ void __launch_bounds__(256)
cgc_layer2(const _Float16* __restrict__ hbuf, const _Float16* __restrict__ w2t,
           const float* __restrict__ bspec, const float* __restrict__ bsh,
           _Float16* __restrict__ eo)
{
    __shared__ _Float16 Bs[2 * LDSB_ELEMS];
    const int e  = (int)blockIdx.z;
    const int w  = (int)(threadIdx.x >> 5);
    const int m0 = (int)blockIdx.y * 128 + w * 16;
    const int n0 = (int)blockIdx.x * 64;
    const _Float16* A    = hbuf + (size_t)e * Bsz * Hh;
    const _Float16* Bt   = w2t + (size_t)e * Ee * Hh;
    const float*    bias = (e < 12) ? (bspec + (size_t)e * Ee)
                                    : (bsh + (size_t)(e - 12) * Ee);
    _Float16* C = eo + (size_t)e * Bsz * Ee;
    cgc_gemm_tile(A, Bt, bias, C, Bs, Hh, Ee, m0, n0);
}

// ---------------------------------------------------------------------------
// Task gates: one wave32 per (t,b) row. 8 logits, softmax, combine 8 experts.
// ---------------------------------------------------------------------------
__global__ void __launch_bounds__(256)
cgc_task_gate_combine(const float* __restrict__ xt, const float* __restrict__ Wg,
                      const float* __restrict__ bg, const _Float16* __restrict__ eo,
                      float* __restrict__ out)
{
    const int lane = (int)(threadIdx.x & 31u);
    const int row  = (int)blockIdx.x * 8 + (int)(threadIdx.x >> 5);  // [0, T*B)
    const int t = row / Bsz, b = row % Bsz;
    const float* x = xt + ((size_t)t * Bsz + b) * Dd;

    float g[8];
#pragma unroll
    for (int j = 0; j < 8; ++j) g[j] = 0.f;
    for (int d = lane; d < Dd; d += 32) {
        const float  xv = x[d];
        const float* wg = Wg + ((size_t)t * Dd + d) * 8;
#pragma unroll
        for (int j = 0; j < 8; ++j) g[j] += xv * wg[j];
    }
#pragma unroll
    for (int j = 0; j < 8; ++j) {
        for (int off = 16; off > 0; off >>= 1) g[j] += __shfl_xor(g[j], off, 32);
        g[j] += bg[t * 8 + j];
    }
    float mx = g[0];
#pragma unroll
    for (int j = 1; j < 8; ++j) mx = fmaxf(mx, g[j]);
    float s = 0.f;
#pragma unroll
    for (int j = 0; j < 8; ++j) { g[j] = __expf(g[j] - mx); s += g[j]; }
    const float inv = 1.f / s;
#pragma unroll
    for (int j = 0; j < 8; ++j) g[j] *= inv;

    const size_t be0 = (size_t)b * Ee;
    for (int c = lane; c < Ee; c += 32) {
        float v = 0.f;
#pragma unroll
        for (int s4 = 0; s4 < 4; ++s4)
            v += g[s4] * (float)eo[(size_t)(t * 4 + s4) * Bsz * Ee + be0 + c];
#pragma unroll
        for (int j = 0; j < 4; ++j)
            v += g[4 + j] * (float)eo[(size_t)(12 + j) * Bsz * Ee + be0 + c];
        out[(size_t)t * Bsz * Ee + be0 + c] = v;
    }
}

// ---------------------------------------------------------------------------
// Shared gate: one wave32 per row b; 16 logits over all experts (task-major
// specific 0..11, then shared 12..15) -> output row index T (=3).
// ---------------------------------------------------------------------------
__global__ void __launch_bounds__(256)
cgc_shared_gate_combine(const float* __restrict__ xs, const float* __restrict__ Wg,
                        const float* __restrict__ bg, const _Float16* __restrict__ eo,
                        float* __restrict__ out)
{
    const int lane = (int)(threadIdx.x & 31u);
    const int b    = (int)blockIdx.x * 8 + (int)(threadIdx.x >> 5);
    const float* x = xs + (size_t)b * Dd;

    float g[16];
#pragma unroll
    for (int j = 0; j < 16; ++j) g[j] = 0.f;
    for (int d = lane; d < Dd; d += 32) {
        const float  xv = x[d];
        const float* wg = Wg + (size_t)d * 16;
#pragma unroll
        for (int j = 0; j < 16; ++j) g[j] += xv * wg[j];
    }
#pragma unroll
    for (int j = 0; j < 16; ++j) {
        for (int off = 16; off > 0; off >>= 1) g[j] += __shfl_xor(g[j], off, 32);
        g[j] += bg[j];
    }
    float mx = g[0];
#pragma unroll
    for (int j = 1; j < 16; ++j) mx = fmaxf(mx, g[j]);
    float s = 0.f;
#pragma unroll
    for (int j = 0; j < 16; ++j) { g[j] = __expf(g[j] - mx); s += g[j]; }
    const float inv = 1.f / s;

    const size_t be0 = (size_t)b * Ee;
    for (int c = lane; c < Ee; c += 32) {
        float v = 0.f;
#pragma unroll
        for (int j = 0; j < 16; ++j)
            v += g[j] * inv * (float)eo[(size_t)j * Bsz * Ee + be0 + c];
        out[(size_t)T * Bsz * Ee + be0 + c] = v;
    }
}

// ---------------------------------------------------------------------------
extern "C" void kernel_launch(void* const* d_in, const int* in_sizes, int n_in,
                              void* d_out, int out_size, void* d_ws, size_t ws_size,
                              hipStream_t stream)
{
    (void)in_sizes; (void)n_in; (void)out_size; (void)ws_size;
    const float* x_tasks   = (const float*)d_in[0];
    const float* x_shared  = (const float*)d_in[1];
    const float* W_spec1   = (const float*)d_in[2];
    const float* b_spec1   = (const float*)d_in[3];
    const float* W_spec2   = (const float*)d_in[4];
    const float* b_spec2   = (const float*)d_in[5];
    const float* W_sh1     = (const float*)d_in[6];
    const float* b_sh1     = (const float*)d_in[7];
    const float* W_sh2     = (const float*)d_in[8];
    const float* b_sh2     = (const float*)d_in[9];
    const float* W_gate    = (const float*)d_in[10];
    const float* b_gate    = (const float*)d_in[11];
    const float* W_gate_sh = (const float*)d_in[12];
    const float* b_gate_sh = (const float*)d_in[13];
    float* out = (float*)d_out;

    // Workspace carve-out (~276 MiB total), 256B aligned slices.
    char*  ws  = (char*)d_ws;
    size_t off = 0;
    auto alloc = [&](size_t bytes) -> void* {
        void* p = ws + off;
        off += (bytes + 255) & ~(size_t)255;
        return p;
    };
    _Float16* xt16 = (_Float16*)alloc((size_t)T * Bsz * Dd * 2);
    _Float16* xs16 = (_Float16*)alloc((size_t)Bsz * Dd * 2);
    _Float16* w1t  = (_Float16*)alloc((size_t)NEXP * Hh * Dd * 2);
    _Float16* w2t  = (_Float16*)alloc((size_t)NEXP * Ee * Hh * 2);
    _Float16* hbuf = (_Float16*)alloc((size_t)NEXP * Bsz * Hh * 2);
    _Float16* eo   = (_Float16*)alloc((size_t)NEXP * Bsz * Ee * 2);

    // 1) precision conversion / weight transposes
    { size_t n = (size_t)T * Bsz * Dd;
      cgc_f32_to_f16<<<dim3((unsigned)(n / 256)), dim3(256), 0, stream>>>(x_tasks, xt16, n); }
    { size_t n = (size_t)Bsz * Dd;
      cgc_f32_to_f16<<<dim3((unsigned)(n / 256)), dim3(256), 0, stream>>>(x_shared, xs16, n); }
    { size_t n = (size_t)NEXP * Dd * Hh;
      cgc_w_transpose<<<dim3((unsigned)(n / 256)), dim3(256), 0, stream>>>(
          W_spec1, W_sh1, 12, w1t, Dd, Hh); }
    { size_t n = (size_t)NEXP * Hh * Ee;
      cgc_w_transpose<<<dim3((unsigned)(n / 256)), dim3(256), 0, stream>>>(
          W_spec2, W_sh2, 12, w2t, Hh, Ee); }

    // 2) expert MLPs on the WMMA pipe (B tiles via async LDS double-buffering)
    cgc_layer1<<<dim3(Hh / 64, Bsz / 128, NEXP), dim3(256), 0, stream>>>(
        xt16, xs16, w1t, b_spec1, b_sh1, hbuf);
    cgc_layer2<<<dim3(Ee / 64, Bsz / 128, NEXP), dim3(256), 0, stream>>>(
        hbuf, w2t, b_spec2, b_sh2, eo);

    // 3) gates + weighted combines (bandwidth-bound, VALU)
    cgc_task_gate_combine<<<dim3(T * Bsz / 8), dim3(256), 0, stream>>>(
        x_tasks, W_gate, b_gate, eo, out);
    cgc_shared_gate_combine<<<dim3(Bsz / 8), dim3(256), 0, stream>>>(
        x_shared, W_gate_sh, b_gate_sh, eo, out);
}
